// ParallelRNN_63230508532270
// MI455X (gfx1250) — matrix operand order repaired and checked by reading, not compiled, not run on hardware
//
#include <hip/hip_runtime.h>
#include <hip/hip_bf16.h>

// ---------------------------------------------------------------------------
// ParallelRNN (per-latent-dim GRU + MLP) for MI455X / gfx1250, wave32 WMMA.
//
// v3: weight B-tiles have ZERO intra-block reuse (each wave consumes its own
// 32 columns), so the LDS staging round-trip for weights is pure overhead.
// A pre-pass converts/transposes all weights to f16 column-major (L,N,K) in
// d_ws (23 MB, L2-resident); the main kernel then loads B fragments directly
// from global as contiguous global_load_b128 pairs in native WMMA layout —
// no weight LDS, no per-K-step barriers, halved L2 weight bytes.
// Falls back to the v2 staged-LDS kernel if ws_size is too small.
// ---------------------------------------------------------------------------

typedef __attribute__((ext_vector_type(16))) _Float16 v16h;
typedef __attribute__((ext_vector_type(8)))  _Float16 v8h;
typedef __attribute__((ext_vector_type(8)))  float    v8f;

union V16 { v16h v; v8h h[2]; };

#define B_ 4096
#define H_ 256
#define L_ 32
#define I_ 64

// LDS strides (halfs), 16B-aligned rows
#define AROW 272                 // A tiles: 32 x 256 f16, row stride 272
#define XROW 72                  // masked-x tile: 32 x 64 f16
#define BROW 40                  // (fallback) weight tile per-column stride
#define WBUF (H_ * BROW)         // (fallback) one weight K-tile buffer

__device__ __forceinline__ v8f wmma16(const v16h& a, const v16h& b, v8f c) {
  // D = A(16x32 f16) * B(32x16 f16) + C(16x16 f32)
  return __builtin_amdgcn_wmma_f32_16x16x32_f16(
      false, a, false, b, (short)0, c, false, false);
}

__device__ __forceinline__ float sigmoidf_(float x) {
  return 1.0f / (1.0f + __expf(-x));
}

// ===========================================================================
// Pre-pass: W (S slices of KxN f32 row-major) -> Wt (S slices of NxK f16)
// ===========================================================================
__global__ __launch_bounds__(256)
void transpose_f16_kernel(const float* __restrict__ W, _Float16* __restrict__ Wt,
                          int K, int N) {
  __shared__ _Float16 tile[64][65];
  const int s  = blockIdx.z;
  const int nb = blockIdx.x * 64;
  const int kb = blockIdx.y * 64;
  const int t  = threadIdx.x;
  const int tr = t >> 6;   // 0..3
  const int tc = t & 63;
  const float*  Ws  = W  + (size_t)s * K * N;
  _Float16*     Wts = Wt + (size_t)s * K * N;
#pragma unroll 4
  for (int it = 0; it < 16; ++it) {
    int r = it * 4 + tr;
    tile[r][tc] = (_Float16)Ws[(size_t)(kb + r) * N + nb + tc];
  }
  __syncthreads();
#pragma unroll 4
  for (int it = 0; it < 16; ++it) {
    int r = it * 4 + tr;
    Wts[(size_t)(nb + r) * K + kb + tc] = tile[tc][r];
  }
}

// ===========================================================================
// v3 GEMM: A(32xK f16, LDS) x Wt(K x 256, f16 stored column-major: col*K + k)
// B fragments loaded directly from global (L2-resident), no barriers.
// ===========================================================================
__device__ __forceinline__ void gemm_tf16(const _Float16* __restrict__ As, int arow,
                                          const _Float16* __restrict__ Wt, int K,
                                          v8f c[4]) {
  const int lane  = threadIdx.x & 31;
  const int wave  = threadIdx.x >> 5;
  const int rA    = lane & 15;
  const int koffA = (lane < 16) ? 0 : 8;   // A: lanes 0-15 hold K 0..7,16..23
  const int koffB = (lane < 16) ? 0 : 8;   // B: lanes 0-15 hold K 0..7,16..23
  const _Float16* Wc0 = Wt + (size_t)(wave * 32 + (lane & 15)) * K;  // column n0
  const _Float16* Wc1 = Wc0 + (size_t)16 * K;                        // column n0+16

  for (int k0 = 0; k0 < K; k0 += 32) {
    V16 a0, a1, b0, b1;
    a0.h[0] = *(const v8h*)&As[rA * arow + k0 + koffA];
    a0.h[1] = *(const v8h*)&As[rA * arow + k0 + koffA + 16];
    a1.h[0] = *(const v8h*)&As[(rA + 16) * arow + k0 + koffA];
    a1.h[1] = *(const v8h*)&As[(rA + 16) * arow + k0 + koffA + 16];
    b0.h[0] = *(const v8h*)&Wc0[k0 + koffB];
    b0.h[1] = *(const v8h*)&Wc0[k0 + koffB + 16];
    b1.h[0] = *(const v8h*)&Wc1[k0 + koffB];
    b1.h[1] = *(const v8h*)&Wc1[k0 + koffB + 16];
    c[0] = wmma16(a0.v, b0.v, c[0]);
    c[1] = wmma16(a0.v, b1.v, c[1]);
    c[2] = wmma16(a1.v, b0.v, c[2]);
    c[3] = wmma16(a1.v, b1.v, c[3]);
  }
}

__global__ __launch_bounds__(256)
void parallel_rnn_f16w_kernel(const float* __restrict__ hidden,
                              const float* __restrict__ x,
                              const float* __restrict__ mask,
                              const _Float16* __restrict__ tWir,
                              const _Float16* __restrict__ tWiz,
                              const _Float16* __restrict__ tWin,
                              const _Float16* __restrict__ tWhr,
                              const _Float16* __restrict__ tWhz,
                              const _Float16* __restrict__ tWhn,
                              const _Float16* __restrict__ tW1,
                              const _Float16* __restrict__ tW2,
                              const float* __restrict__ bir, const float* __restrict__ biz,
                              const float* __restrict__ bin_, const float* __restrict__ bhn,
                              const float* __restrict__ b1,  const float* __restrict__ b2,
                              const float* __restrict__ Wmu, const float* __restrict__ bmu,
                              const float* __restrict__ log_var,
                              float* __restrict__ outH2, float* __restrict__ outMu,
                              float* __restrict__ outLv)
{
  __shared__ __align__(16) _Float16 Ah[32 * AROW];   // h / hnew / h1 (17.4 KB)
  __shared__ __align__(16) _Float16 Ax[32 * XROW];   // masked-x (4.6 KB)
  __shared__ __align__(16) float    h2f[32 * H_];    // h2 scratch (32 KB)
  __shared__ float part[32][9];

  const int t    = threadIdx.x;
  const int lane = t & 31;
  const int wave = t >> 5;
  const int l    = blockIdx.y;
  const int b0   = blockIdx.x * 32;

  // ---- Stage A tiles (f32 -> f16) ------------------------------------------
#pragma unroll 4
  for (int m = 0; m < 32; ++m)
    Ah[m * AROW + t] = (_Float16)hidden[((size_t)(b0 + m) * H_ + t) * L_ + l];
#pragma unroll
  for (int rep = 0; rep < 8; ++rep) {
    int idx = rep * 256 + t;
    int m = idx >> 6, i = idx & 63;
    Ax[m * XROW + i] = (_Float16)(x[(size_t)(b0 + m) * I_ + i] * mask[i * L_ + l]);
  }
  __syncthreads();

  // ---- GRU gate GEMMs (no LDS writes, no barriers) -------------------------
  v8f cr[4] = {}, cz[4] = {}, cx[4] = {}, ch[4] = {};
  const size_t wi = (size_t)l * I_ * H_;
  const size_t wh = (size_t)l * H_ * H_;

  gemm_tf16(Ax, XROW, tWir + wi, I_, cr);   // x·Wir
  gemm_tf16(Ah, AROW, tWhr + wh, H_, cr);   //  + h·Whr
  gemm_tf16(Ax, XROW, tWiz + wi, I_, cz);
  gemm_tf16(Ah, AROW, tWhz + wh, H_, cz);
  gemm_tf16(Ax, XROW, tWin + wi, I_, cx);   // x·Win (separate: r gates hh only)
  gemm_tf16(Ah, AROW, tWhn + wh, H_, ch);   // h·Whn

  // ---- GRU elementwise -----------------------------------------------------
  const int col0 = wave * 32 + (lane & 15);
  const int col1 = col0 + 16;
  const int mofs = (lane < 16) ? 0 : 8;

  float hn[4][8];
  {
    const float br0 = bir[l * H_ + col0],  br1 = bir[l * H_ + col1];
    const float bz0 = biz[l * H_ + col0],  bz1 = biz[l * H_ + col1];
    const float bn0 = bin_[l * H_ + col0], bn1 = bin_[l * H_ + col1];
    const float bh0 = bhn[l * H_ + col0],  bh1 = bhn[l * H_ + col1];
#pragma unroll
    for (int mt = 0; mt < 2; ++mt)
#pragma unroll
      for (int j = 0; j < 8; ++j) {
        const int m = mt * 16 + j + mofs;
        const int f0 = mt * 2, f1 = mt * 2 + 1;
        float r0 = sigmoidf_(cr[f0][j] + br0);
        float z0 = sigmoidf_(cz[f0][j] + bz0);
        float n0 = tanhf(cx[f0][j] + bn0 + r0 * (ch[f0][j] + bh0));
        float h0 = (float)Ah[m * AROW + col0];
        hn[f0][j] = (1.0f - z0) * n0 + z0 * h0;

        float r1 = sigmoidf_(cr[f1][j] + br1);
        float z1 = sigmoidf_(cz[f1][j] + bz1);
        float n1 = tanhf(cx[f1][j] + bn1 + r1 * (ch[f1][j] + bh1));
        float h1v = (float)Ah[m * AROW + col1];
        hn[f1][j] = (1.0f - z1) * n1 + z1 * h1v;
      }
  }
  __syncthreads();  // all waves done reading old Ah (gemms + h mix)
#pragma unroll
  for (int mt = 0; mt < 2; ++mt)
#pragma unroll
    for (int j = 0; j < 8; ++j) {
      const int m = mt * 16 + j + mofs;
      Ah[m * AROW + col0] = (_Float16)hn[mt * 2][j];
      Ah[m * AROW + col1] = (_Float16)hn[mt * 2 + 1][j];
    }
  __syncthreads();

  // ---- MLP layer 1 ---------------------------------------------------------
  {
    v8f d[4] = {};
    gemm_tf16(Ah, AROW, tW1 + wh, H_, d);
    const float bb0 = b1[l * H_ + col0], bb1 = b1[l * H_ + col1];
#pragma unroll
    for (int mt = 0; mt < 2; ++mt)
#pragma unroll
      for (int j = 0; j < 8; ++j) {
        hn[mt * 2][j]     = fmaxf(d[mt * 2][j] + bb0, 0.0f);
        hn[mt * 2 + 1][j] = fmaxf(d[mt * 2 + 1][j] + bb1, 0.0f);
      }
    __syncthreads();  // all waves done reading hnew in Ah
#pragma unroll
    for (int mt = 0; mt < 2; ++mt)
#pragma unroll
      for (int j = 0; j < 8; ++j) {
        const int m = mt * 16 + j + mofs;
        Ah[m * AROW + col0] = (_Float16)hn[mt * 2][j];
        Ah[m * AROW + col1] = (_Float16)hn[mt * 2 + 1][j];
      }
    __syncthreads();
  }

  // ---- MLP layer 2 + outputs ----------------------------------------------
  {
    v8f e[4] = {};
    gemm_tf16(Ah, AROW, tW2 + wh, H_, e);
    const float bb0 = b2[l * H_ + col0], bb1 = b2[l * H_ + col1];
#pragma unroll
    for (int mt = 0; mt < 2; ++mt)
#pragma unroll
      for (int j = 0; j < 8; ++j) {
        const int m = mt * 16 + j + mofs;
        float v0 = fmaxf(e[mt * 2][j] + bb0, 0.0f);
        float v1 = fmaxf(e[mt * 2 + 1][j] + bb1, 0.0f);
        outH2[((size_t)(b0 + m) * H_ + col0) * L_ + l] = v0;
        outH2[((size_t)(b0 + m) * H_ + col1) * L_ + l] = v1;
        h2f[m * H_ + col0] = v0;
        h2f[m * H_ + col1] = v1;
      }
  }
  __syncthreads();

  // ---- mu / lv -------------------------------------------------------------
  {
    const int row = t >> 3, seg = t & 7;
    float s = 0.0f;
#pragma unroll 8
    for (int j = 0; j < 32; ++j)
      s += h2f[row * H_ + seg * 32 + j] * Wmu[l * H_ + seg * 32 + j];
    part[row][seg] = s;
  }
  __syncthreads();
  if (t < 32) {
    float s = bmu[l];
#pragma unroll
    for (int j = 0; j < 8; ++j) s += part[t][j];
    outMu[(size_t)(b0 + t) * L_ + l] = s;
    outLv[(size_t)(b0 + t) * L_ + l] = fmaxf(log_var[l], -3.0f);
  }
}

// ===========================================================================
// Fallback (v2): f32 weights staged through double-buffered LDS
// ===========================================================================
__device__ __forceinline__ void stageW(_Float16* __restrict__ Wb,
                                       const float* __restrict__ Wg, int k0) {
  const int t = threadIdx.x;
#pragma unroll 8
  for (int kk = 0; kk < 32; ++kk)
    Wb[t * BROW + kk] = (_Float16)Wg[(size_t)(k0 + kk) * H_ + t];
}

__device__ __forceinline__ void gemm_nt(const _Float16* __restrict__ As, int arow,
                                        const float* __restrict__ Wg, int K,
                                        _Float16* __restrict__ Wl, v8f c[4]) {
  const int lane  = threadIdx.x & 31;
  const int wave  = threadIdx.x >> 5;
  const int rA    = lane & 15;
  const int koffA = (lane < 16) ? 0 : 8;
  const int koffB = (lane < 16) ? 0 : 8;
  const int cB0   = wave * 32 + (lane & 15);
  const int cB1   = cB0 + 16;

  stageW(Wl, Wg, 0);
  __syncthreads();
  int buf = 0;
  for (int k0 = 0; k0 < K; k0 += 32, buf ^= 1) {
    if (k0 + 32 < K) stageW(Wl + (buf ^ 1) * WBUF, Wg, k0 + 32);
    const _Float16* Wc = Wl + buf * WBUF;
    V16 a0, a1, b0, b1;
    a0.h[0] = *(const v8h*)&As[rA * arow + k0 + koffA];
    a0.h[1] = *(const v8h*)&As[rA * arow + k0 + koffA + 16];
    a1.h[0] = *(const v8h*)&As[(rA + 16) * arow + k0 + koffA];
    a1.h[1] = *(const v8h*)&As[(rA + 16) * arow + k0 + koffA + 16];
    b0.h[0] = *(const v8h*)&Wc[cB0 * BROW + koffB];
    b0.h[1] = *(const v8h*)&Wc[cB0 * BROW + koffB + 16];
    b1.h[0] = *(const v8h*)&Wc[cB1 * BROW + koffB];
    b1.h[1] = *(const v8h*)&Wc[cB1 * BROW + koffB + 16];
    c[0] = wmma16(a0.v, b0.v, c[0]);
    c[1] = wmma16(a0.v, b1.v, c[1]);
    c[2] = wmma16(a1.v, b0.v, c[2]);
    c[3] = wmma16(a1.v, b1.v, c[3]);
    __syncthreads();
  }
}

__global__ __launch_bounds__(256)
void parallel_rnn_f32w_kernel(const float* __restrict__ hidden,
                              const float* __restrict__ x,
                              const float* __restrict__ mask,
                              const float* __restrict__ Wir, const float* __restrict__ bir,
                              const float* __restrict__ Wiz, const float* __restrict__ biz,
                              const float* __restrict__ Win, const float* __restrict__ bin_,
                              const float* __restrict__ Whr,
                              const float* __restrict__ Whz,
                              const float* __restrict__ Whn, const float* __restrict__ bhn,
                              const float* __restrict__ W1,  const float* __restrict__ b1,
                              const float* __restrict__ W2,  const float* __restrict__ b2,
                              const float* __restrict__ Wmu, const float* __restrict__ bmu,
                              const float* __restrict__ log_var,
                              float* __restrict__ outH2, float* __restrict__ outMu,
                              float* __restrict__ outLv)
{
  __shared__ __align__(16) _Float16 Ah[32 * AROW];
  __shared__ __align__(16) _Float16 Ax[32 * XROW];
  __shared__ __align__(16) _Float16 Wl[2 * WBUF];
  __shared__ float part[32][9];

  const int t    = threadIdx.x;
  const int lane = t & 31;
  const int wave = t >> 5;
  const int l    = blockIdx.y;
  const int b0   = blockIdx.x * 32;

#pragma unroll 4
  for (int m = 0; m < 32; ++m)
    Ah[m * AROW + t] = (_Float16)hidden[((size_t)(b0 + m) * H_ + t) * L_ + l];
#pragma unroll
  for (int rep = 0; rep < 8; ++rep) {
    int idx = rep * 256 + t;
    int m = idx >> 6, i = idx & 63;
    Ax[m * XROW + i] = (_Float16)(x[(size_t)(b0 + m) * I_ + i] * mask[i * L_ + l]);
  }
  __syncthreads();

  v8f cr[4] = {}, cz[4] = {}, cx[4] = {}, ch[4] = {};
  const size_t wi = (size_t)l * I_ * H_;
  const size_t wh = (size_t)l * H_ * H_;

  gemm_nt(Ax, XROW, Wir + wi, I_, Wl, cr);
  gemm_nt(Ah, AROW, Whr + wh, H_, Wl, cr);
  gemm_nt(Ax, XROW, Wiz + wi, I_, Wl, cz);
  gemm_nt(Ah, AROW, Whz + wh, H_, Wl, cz);
  gemm_nt(Ax, XROW, Win + wi, I_, Wl, cx);
  gemm_nt(Ah, AROW, Whn + wh, H_, Wl, ch);

  const int col0 = wave * 32 + (lane & 15);
  const int col1 = col0 + 16;
  const int mofs = (lane < 16) ? 0 : 8;

  float hn[4][8];
  {
    const float br0 = bir[l * H_ + col0],  br1 = bir[l * H_ + col1];
    const float bz0 = biz[l * H_ + col0],  bz1 = biz[l * H_ + col1];
    const float bn0 = bin_[l * H_ + col0], bn1 = bin_[l * H_ + col1];
    const float bh0 = bhn[l * H_ + col0],  bh1 = bhn[l * H_ + col1];
#pragma unroll
    for (int mt = 0; mt < 2; ++mt)
#pragma unroll
      for (int j = 0; j < 8; ++j) {
        const int m = mt * 16 + j + mofs;
        const int f0 = mt * 2, f1 = mt * 2 + 1;
        float r0 = sigmoidf_(cr[f0][j] + br0);
        float z0 = sigmoidf_(cz[f0][j] + bz0);
        float n0 = tanhf(cx[f0][j] + bn0 + r0 * (ch[f0][j] + bh0));
        float h0 = (float)Ah[m * AROW + col0];
        hn[f0][j] = (1.0f - z0) * n0 + z0 * h0;
        float r1 = sigmoidf_(cr[f1][j] + br1);
        float z1 = sigmoidf_(cz[f1][j] + bz1);
        float n1 = tanhf(cx[f1][j] + bn1 + r1 * (ch[f1][j] + bh1));
        float h1v = (float)Ah[m * AROW + col1];
        hn[f1][j] = (1.0f - z1) * n1 + z1 * h1v;
      }
  }
  __syncthreads();
#pragma unroll
  for (int mt = 0; mt < 2; ++mt)
#pragma unroll
    for (int j = 0; j < 8; ++j) {
      const int m = mt * 16 + j + mofs;
      Ah[m * AROW + col0] = (_Float16)hn[mt * 2][j];
      Ah[m * AROW + col1] = (_Float16)hn[mt * 2 + 1][j];
    }
  __syncthreads();

  {
    v8f d[4] = {};
    gemm_nt(Ah, AROW, W1 + wh, H_, Wl, d);
    const float bb0 = b1[l * H_ + col0], bb1 = b1[l * H_ + col1];
#pragma unroll
    for (int mt = 0; mt < 2; ++mt)
#pragma unroll
      for (int j = 0; j < 8; ++j) {
        hn[mt * 2][j]     = fmaxf(d[mt * 2][j] + bb0, 0.0f);
        hn[mt * 2 + 1][j] = fmaxf(d[mt * 2 + 1][j] + bb1, 0.0f);
      }
    __syncthreads();
#pragma unroll
    for (int mt = 0; mt < 2; ++mt)
#pragma unroll
      for (int j = 0; j < 8; ++j) {
        const int m = mt * 16 + j + mofs;
        Ah[m * AROW + col0] = (_Float16)hn[mt * 2][j];
        Ah[m * AROW + col1] = (_Float16)hn[mt * 2 + 1][j];
      }
    __syncthreads();
  }

  v8f e[4] = {};
  gemm_nt(Ah, AROW, W2 + wh, H_, Wl, e);

  float* h2f = (float*)Wl;
  {
    const float bb0 = b2[l * H_ + col0], bb1 = b2[l * H_ + col1];
#pragma unroll
    for (int mt = 0; mt < 2; ++mt)
#pragma unroll
      for (int j = 0; j < 8; ++j) {
        const int m = mt * 16 + j + mofs;
        float v0 = fmaxf(e[mt * 2][j] + bb0, 0.0f);
        float v1 = fmaxf(e[mt * 2 + 1][j] + bb1, 0.0f);
        outH2[((size_t)(b0 + m) * H_ + col0) * L_ + l] = v0;
        outH2[((size_t)(b0 + m) * H_ + col1) * L_ + l] = v1;
        h2f[m * H_ + col0] = v0;
        h2f[m * H_ + col1] = v1;
      }
  }
  __syncthreads();

  {
    const int row = t >> 3, seg = t & 7;
    float s = 0.0f;
#pragma unroll 8
    for (int j = 0; j < 32; ++j)
      s += h2f[row * H_ + seg * 32 + j] * Wmu[l * H_ + seg * 32 + j];
    part[row][seg] = s;
  }
  __syncthreads();
  if (t < 32) {
    float s = bmu[l];
#pragma unroll
    for (int j = 0; j < 8; ++j) s += part[t][j];
    outMu[(size_t)(b0 + t) * L_ + l] = s;
    outLv[(size_t)(b0 + t) * L_ + l] = fmaxf(log_var[l], -3.0f);
  }
}

// ===========================================================================
extern "C" void kernel_launch(void* const* d_in, const int* in_sizes, int n_in,
                              void* d_out, int out_size, void* d_ws, size_t ws_size,
                              hipStream_t stream) {
  const float* hidden = (const float*)d_in[0];
  const float* x      = (const float*)d_in[1];
  const float* mask   = (const float*)d_in[2];
  const float* Wir    = (const float*)d_in[3];
  const float* bir    = (const float*)d_in[4];
  const float* Wiz    = (const float*)d_in[5];
  const float* biz    = (const float*)d_in[6];
  const float* Win    = (const float*)d_in[7];
  const float* bin_   = (const float*)d_in[8];
  const float* Whr    = (const float*)d_in[9];
  const float* Whz    = (const float*)d_in[10];
  const float* Whn    = (const float*)d_in[11];
  const float* bhn    = (const float*)d_in[12];
  const float* W1     = (const float*)d_in[13];
  const float* b1     = (const float*)d_in[14];
  const float* W2     = (const float*)d_in[15];
  const float* b2     = (const float*)d_in[16];
  const float* Wmu    = (const float*)d_in[17];
  const float* bmu    = (const float*)d_in[18];
  const float* lv     = (const float*)d_in[19];

  float* outH2 = (float*)d_out;
  float* outMu = outH2 + (size_t)B_ * H_ * L_;
  float* outLv = outMu + (size_t)B_ * L_;

  const size_t szH = (size_t)L_ * H_ * H_;   // halfs per hidden tensor
  const size_t szI = (size_t)L_ * I_ * H_;   // halfs per input tensor
  const size_t needBytes = (5 * szH + 3 * szI) * sizeof(_Float16);

  dim3 grid(B_ / 32, L_);
  dim3 block(256);

  if (ws_size >= needBytes) {
    _Float16* w    = (_Float16*)d_ws;
    _Float16* tWhr = w;              w += szH;
    _Float16* tWhz = w;              w += szH;
    _Float16* tWhn = w;              w += szH;
    _Float16* tW1  = w;              w += szH;
    _Float16* tW2  = w;              w += szH;
    _Float16* tWir = w;              w += szI;
    _Float16* tWiz = w;              w += szI;
    _Float16* tWin = w;              w += szI;

    dim3 gH(H_ / 64, H_ / 64, L_);   // hidden weights: 256x256 per slice
    dim3 gI(H_ / 64, I_ / 64, L_);   // input weights:  64x256 per slice
    transpose_f16_kernel<<<gH, block, 0, stream>>>(Whr, tWhr, H_, H_);
    transpose_f16_kernel<<<gH, block, 0, stream>>>(Whz, tWhz, H_, H_);
    transpose_f16_kernel<<<gH, block, 0, stream>>>(Whn, tWhn, H_, H_);
    transpose_f16_kernel<<<gH, block, 0, stream>>>(W1,  tW1,  H_, H_);
    transpose_f16_kernel<<<gH, block, 0, stream>>>(W2,  tW2,  H_, H_);
    transpose_f16_kernel<<<gI, block, 0, stream>>>(Wir, tWir, I_, H_);
    transpose_f16_kernel<<<gI, block, 0, stream>>>(Wiz, tWiz, I_, H_);
    transpose_f16_kernel<<<gI, block, 0, stream>>>(Win, tWin, I_, H_);

    parallel_rnn_f16w_kernel<<<grid, block, 0, stream>>>(
        hidden, x, mask, tWir, tWiz, tWin, tWhr, tWhz, tWhn, tW1, tW2,
        bir, biz, bin_, bhn, b1, b2, Wmu, bmu, lv, outH2, outMu, outLv);
  } else {
    parallel_rnn_f32w_kernel<<<grid, block, 0, stream>>>(
        hidden, x, mask, Wir, bir, Wiz, biz, Win, bin_,
        Whr, Whz, Whn, bhn, W1, b1, W2, b2, Wmu, bmu, lv,
        outH2, outMu, outLv);
  }
}
